// VQVAE_48524540511009
// MI455X (gfx1250) — matrix operand order repaired and checked by reading, not compiled, not run on hardware
//
#include <hip/hip_runtime.h>

// ---------------------------------------------------------------------------
// Types for CDNA5 WMMA
// ---------------------------------------------------------------------------
typedef __attribute__((ext_vector_type(16))) __bf16 v16bf;
typedef __attribute__((ext_vector_type(8)))  float  v8f;

union BFrag { uint4 q[2]; v16bf v; };

__device__ __forceinline__ unsigned short f2bf(float f) {
    union { float f; unsigned u; } x; x.f = f;
    unsigned u = x.u;
    unsigned r = u + 0x7FFFu + ((u >> 16) & 1u);   // round-to-nearest-even
    return (unsigned short)(r >> 16);
}

// ---------------------------------------------------------------------------
// Problem constants
// ---------------------------------------------------------------------------
#define BB     256          // batch
#define CH     256          // hidden / embedding dim
#define NCODE  8192         // codebook size
#define TQ     64           // latent length (512 / 8)
#define NROWS  (BB * TQ)    // 16384 flattened latent rows

// ---------------------------------------------------------------------------
// 1) f32 -> bf16 copies (embedding table / raw input x)
// ---------------------------------------------------------------------------
__global__ __launch_bounds__(256) void tobf16_kernel(const float* __restrict__ src,
                                                     unsigned short* __restrict__ dst) {
    int t = blockIdx.x * 256 + threadIdx.x;
    dst[t] = f2bf(src[t]);
}

// 2) Per-code squared norms (f32, exact)
__global__ __launch_bounds__(256) void norms_kernel(const float* __restrict__ emb,
                                                    float* __restrict__ norms) {
    int k = blockIdx.x * 256 + threadIdx.x;       // NCODE total
    const float4* e = (const float4*)(emb + k * CH);
    float s = 0.f;
    #pragma unroll 8
    for (int i = 0; i < CH / 4; ++i) {
        float4 v = e[i];
        s += v.x * v.x + v.y * v.y + v.z * v.z + v.w * v.w;
    }
    norms[k] = s;
}

// 3) Weight prep: w (Co, Ci, ntaps) f32 -> wbf[(tap*Co + co)*Ci + ci] bf16
//    (co-major, ci-contiguous per tap: WMMA B-operand feed, two 16B runs/lane)
__global__ __launch_bounds__(256) void wprep_kernel(const float* __restrict__ w,
                                                    unsigned short* __restrict__ wbf,
                                                    int Co, int Ci, int ntaps) {
    int t = blockIdx.x * 256 + threadIdx.x;
    if (t >= Co * Ci * ntaps) return;
    int ci = t % Ci; int co = (t / Ci) % Co; int tap = t / (Ci * Co);
    wbf[t] = f2bf(w[(co * Ci + ci) * ntaps + tap]);
}

// ---------------------------------------------------------------------------
// 4) Generic implicit-GEMM conv/deconv/proj kernel (bf16 WMMA, f32 accumulate)
//
//    A   : activations, rows = (b*Li + p), Ci bf16 channels per row
//    W   : (ntaps, Co, Ci) bf16, co-major (ci contiguous)
//    out : rows = (b*Lo_out + out_stride*j + out_off), Co channels
//    y[j, co] = bias[co] + sum_t sum_ci W[wtap_t, co, ci] * A[in_stride*j + off_t, ci]
//    (tap rows outside [0, Li) read as zero -> conv padding for free)
//
//    One 16x16 output tile per wave; no LDS, no barriers; per-lane row gather.
// ---------------------------------------------------------------------------
__global__ __launch_bounds__(256) void conv_wmma_kernel(
        const unsigned short* __restrict__ Abf,
        const unsigned short* __restrict__ Wbf,
        const float* __restrict__ bias,
        float* __restrict__ outF,                 // nullable
        unsigned short* __restrict__ outB,        // nullable
        int Ci, int Co, int Li, int Lo,
        int in_stride, int off0, int off1, int off2, int ntaps,
        int wtap0, int wtap1, int wtap2,
        int Lo_out, int out_stride, int out_off, int relu, int ntiles) {
    const int lane = threadIdx.x & 31;
    const int wave = threadIdx.x >> 5;
    const int col  = lane & 15;
    const int hi   = lane >> 4;
    const int gw   = blockIdx.x * 8 + wave;       // global wave id
    const int cot  = gw / ntiles;
    if (cot * 16 >= Co) return;
    const int tile = gw % ntiles;
    const int co0  = cot * 16;
    const int lpb  = Lo >> 4;                     // 16-position tiles per batch
    const int b    = tile / lpb;
    const int j0   = (tile % lpb) << 4;

    const uint4 Z4 = {0u, 0u, 0u, 0u};
    v8f acc = {};

    for (int t = 0; t < ntaps; ++t) {
        const int off = (t == 0) ? off0 : ((t == 1) ? off1 : off2);
        const int wt  = (t == 0) ? wtap0 : ((t == 1) ? wtap1 : wtap2);
        const int p   = in_stride * (j0 + col) + off;        // per-lane gathered row
        const bool valid = (p >= 0) && (p < Li);
        const unsigned short* arow = Abf + ((long)b * Li + (valid ? p : 0)) * Ci;
        const unsigned short* brow = Wbf + ((long)(wt * Co + co0 + col)) * Ci + hi * 8;
        for (int kb = 0; kb < Ci; kb += 32) {
            BFrag a, bb;
            if (valid) {
                a.q[0] = *(const uint4*)(arow + kb + hi * 8);
                a.q[1] = *(const uint4*)(arow + kb + 16 + hi * 8);
            } else { a.q[0] = Z4; a.q[1] = Z4; }
            bb.q[0] = *(const uint4*)(brow + kb);
            bb.q[1] = *(const uint4*)(brow + kb + 16);
            acc = __builtin_amdgcn_wmma_f32_16x16x32_bf16(
                    false, a.v, false, bb.v, (short)0, acc, false, false);
        }
    }

    const float bs = bias[co0 + col];
    #pragma unroll
    for (int v = 0; v < 8; ++v) {
        float val = acc[v] + bs;
        if (relu) val = fmaxf(val, 0.0f);
        int  j    = j0 + v + 8 * hi;              // D-layout row = v + 8*hi
        long oidx = ((long)b * Lo_out + out_stride * j + out_off) * Co + co0 + col;
        if (outF) outF[oidx] = val;
        if (outB) outB[oidx] = f2bf(val);
    }
}

// ---------------------------------------------------------------------------
// 5) VQ argmin:  dist = ||e||^2 - 2 * (flat . e)   via v_wmma_f32_16x16x32_bf16
//    (unchanged: resident A fragments, LDS-staged 16-code tile per step)
// ---------------------------------------------------------------------------
__global__ __launch_bounds__(128) void vq_argmin_kernel(
        const unsigned short* __restrict__ flatA,   // NROWS x CH bf16
        const unsigned short* __restrict__ embBF,   // NCODE x CH bf16
        const float* __restrict__ norms,            // NCODE
        int* __restrict__ idxOut) {                 // NROWS
    __shared__ alignas(16) unsigned short sB[16 * CH];
    __shared__ float sN[16];

    const int tid  = threadIdx.x;
    const int lane = tid & 31;
    const int wave = tid >> 5;
    const int col  = lane & 15;
    const int hi   = lane >> 4;
    const int m0   = blockIdx.x * 64 + wave * 16;

    BFrag a[8];
    const unsigned short* arow = flatA + (m0 + col) * CH;
    #pragma unroll
    for (int kb = 0; kb < 8; ++kb) {
        int c0 = kb * 32 + hi * 8;
        a[kb].q[0] = *(const uint4*)(arow + c0);
        a[kb].q[1] = *(const uint4*)(arow + c0 + 16);
    }

    float best[8]; int bidx[8];
    #pragma unroll
    for (int v = 0; v < 8; ++v) { best[v] = 3.4e38f; bidx[v] = 0; }

    for (int n0 = 0; n0 < NCODE; n0 += 16) {
        {
            int code = tid >> 3, seg = tid & 7;
            const uint4* src = (const uint4*)(embBF + (n0 + code) * CH + seg * 32);
            uint4* dst = (uint4*)(sB + code * CH + seg * 32);
            dst[0] = src[0]; dst[1] = src[1];
        }
        if (tid < 16) sN[tid] = norms[n0 + tid];
        __syncthreads();

        v8f acc = {};
        const unsigned short* brow = sB + col * CH + hi * 8;
        #pragma unroll
        for (int kb = 0; kb < 8; ++kb) {
            BFrag b;
            b.q[0] = *(const uint4*)(brow + kb * 32);
            b.q[1] = *(const uint4*)(brow + kb * 32 + 16);
            acc = __builtin_amdgcn_wmma_f32_16x16x32_bf16(
                    false, a[kb].v, false, b.v, (short)0, acc, false, false);
        }

        float nn  = sN[col];
        int  code = n0 + col;
        #pragma unroll
        for (int v = 0; v < 8; ++v) {
            float d = nn - 2.0f * acc[v];
            if (d < best[v]) { best[v] = d; bidx[v] = code; }
        }
        __syncthreads();
    }

    #pragma unroll
    for (int off = 1; off < 16; off <<= 1) {
        #pragma unroll
        for (int v = 0; v < 8; ++v) {
            float ov = __shfl_xor(best[v], off, 32);
            int   oi = __shfl_xor(bidx[v], off, 32);
            if (ov < best[v] || (ov == best[v] && oi < bidx[v])) { best[v] = ov; bidx[v] = oi; }
        }
    }
    if (col == 0) {
        #pragma unroll
        for (int v = 0; v < 8; ++v) idxOut[m0 + v + 8 * hi] = bidx[v];
    }
}

// ---------------------------------------------------------------------------
// 6) Gather z_q (bf16 rows for decoder) + per-block partial of sum((z_e-z_q)^2)
//    (exact f32 loss; deterministic block partials, no float atomics)
// ---------------------------------------------------------------------------
__global__ __launch_bounds__(256) void gather_loss_kernel(
        const int* __restrict__ idx, const float* __restrict__ emb,
        const float* __restrict__ zeF, unsigned short* __restrict__ zqBF,
        float* __restrict__ partials) {
    int t = blockIdx.x * 256 + threadIdx.x;       // NROWS*CH, c fastest
    int c = t & 255; int r = t >> 8;
    float q = emb[idx[r] * CH + c];               // coalesced: warp spans 32 c's
    float e = zeF[t];
    zqBF[t] = f2bf(q);
    float d = e - q; d *= d;
    #pragma unroll
    for (int off = 16; off; off >>= 1) d += __shfl_down(d, off, 32);
    __shared__ float sw[8];
    if ((threadIdx.x & 31) == 0) sw[threadIdx.x >> 5] = d;
    __syncthreads();
    if (threadIdx.x == 0) {
        float s = 0.f;
        #pragma unroll
        for (int i = 0; i < 8; ++i) s += sw[i];
        partials[blockIdx.x] = s;
    }
}

// 7) Finalize: deterministic loss reduce (block 0) + index copy
__global__ __launch_bounds__(256) void finalize_kernel(
        const float* __restrict__ partials, const int* __restrict__ idx,
        float* __restrict__ outLoss, int* __restrict__ outIdx) {
    int t = blockIdx.x * 256 + threadIdx.x;
    if (t < NROWS) outIdx[t] = idx[t];
    if (blockIdx.x == 0) {
        __shared__ float sh[256];
        float s = 0.f;
        for (int i = threadIdx.x; i < NROWS; i += 256) s += partials[i];
        sh[threadIdx.x] = s;
        __syncthreads();
        for (int off = 128; off; off >>= 1) {
            if (threadIdx.x < off) sh[threadIdx.x] += sh[threadIdx.x + off];
            __syncthreads();
        }
        if (threadIdx.x == 0) {
            float l = sh[0] * (1.0f / ((float)NROWS * (float)CH));
            outLoss[0] = l;   // codebook_loss (== commitment_loss in fwd math)
            outLoss[1] = l;   // commitment_loss
        }
    }
}

// ---------------------------------------------------------------------------
// Host-side orchestration
// ---------------------------------------------------------------------------
extern "C" void kernel_launch(void* const* d_in, const int* in_sizes, int n_in,
                              void* d_out, int out_size, void* d_ws, size_t ws_size,
                              hipStream_t stream) {
    const float* x      = (const float*)d_in[0];
    const float* enc_w0 = (const float*)d_in[1];  const float* enc_b0 = (const float*)d_in[2];
    const float* enc_w1 = (const float*)d_in[3];  const float* enc_b1 = (const float*)d_in[4];
    const float* enc_w2 = (const float*)d_in[5];  const float* enc_b2 = (const float*)d_in[6];
    const float* proj_w = (const float*)d_in[7];  const float* proj_b = (const float*)d_in[8];
    const float* emb    = (const float*)d_in[9];
    const float* dec_w0 = (const float*)d_in[10]; const float* dec_b0 = (const float*)d_in[11];
    const float* dec_w1 = (const float*)d_in[12]; const float* dec_b1 = (const float*)d_in[13];
    const float* dec_w2 = (const float*)d_in[14]; const float* dec_b2 = (const float*)d_in[15];
    const float* dec_w3 = (const float*)d_in[16]; const float* dec_b3 = (const float*)d_in[17];

    // workspace layout (byte offsets, lifetime-aliased, ~85 MB total)
    char* ws = (char*)d_ws;
    unsigned short* xBF   = (unsigned short*)(ws + 0);          // 8.4 MB (dead after L0)
    unsigned short* d1    = (unsigned short*)(ws + 0);          // 16.8 MB (reuses xBF)
    unsigned short* h0    = (unsigned short*)(ws + 16777216);   // 8.4 MB
    unsigned short* h1    = (unsigned short*)(ws + 25165824);   // 8.4 MB
    unsigned short* d2    = (unsigned short*)(ws + 16777216);   // 16.8 MB (reuses h0+h1)
    unsigned short* h2    = (unsigned short*)(ws + 33554432);   // 8.4 MB (dead after proj)
    unsigned short* zqBF  = (unsigned short*)(ws + 33554432);   // 8.4 MB (reuses h2)
    unsigned short* flatA = (unsigned short*)(ws + 41943040);   // 8.4 MB
    float*          zeF   = (float*)(ws + 50331648);            // 16.8 MB
    unsigned short* d0    = (unsigned short*)(ws + 67108864);   // 16.8 MB
    unsigned short* embBF = (unsigned short*)(ws + 83886080);   // 4.2 MB
    float*          norms = (float*)(ws + 88080384);
    int*            idxB  = (int*)(ws + 88113152);
    float*          part  = (float*)(ws + 88178688);
    unsigned short* wE0   = (unsigned short*)(ws + 88244224);
    unsigned short* wE1   = (unsigned short*)(ws + 88260608);
    unsigned short* wE2   = (unsigned short*)(ws + 88309760);
    unsigned short* wP    = (unsigned short*)(ws + 88506368);
    unsigned short* wD0   = (unsigned short*)(ws + 88637440);
    unsigned short* wD1   = (unsigned short*)(ws + 89030656);
    unsigned short* wD2   = (unsigned short*)(ws + 89227264);
    unsigned short* wD3   = (unsigned short*)(ws + 89276416);

    float* out     = (float*)d_out;                 // (256, 1024, 32) == rows x 32 ch
    float* outLoss = out + 256 * 1024 * 32;
    int*   outIdx  = (int*)(out + 256 * 1024 * 32 + 2);

    auto conv = [&](const unsigned short* A, const unsigned short* W, const float* bs,
                    float* oF, unsigned short* oB, int Ci, int Co, int Li, int Lo,
                    int istr, int o0, int o1, int o2, int nt, int w0, int w1, int w2,
                    int LoOut, int ostr, int ooff, int relu) {
        int ntiles = BB * Lo / 16;
        int waves  = ntiles * (Co / 16);            // 16384 for every layer here
        conv_wmma_kernel<<<waves / 8, 256, 0, stream>>>(
            A, W, bs, oF, oB, Ci, Co, Li, Lo, istr, o0, o1, o2, nt,
            w0, w1, w2, LoOut, ostr, ooff, relu, ntiles);
    };

    // ---- prep: bf16 copies, weight transposes, codebook norms ----------
    tobf16_kernel<<<NCODE * CH / 256, 256, 0, stream>>>(emb, embBF);
    tobf16_kernel<<<BB * 512 * 32 / 256, 256, 0, stream>>>(x, xBF);
    norms_kernel<<<NCODE / 256, 256, 0, stream>>>(emb, norms);
    wprep_kernel<<<(64 * 32 * 3 + 255) / 256, 256, 0, stream>>>(enc_w0, wE0, 64, 32, 3);
    wprep_kernel<<<(128 * 64 * 3 + 255) / 256, 256, 0, stream>>>(enc_w1, wE1, 128, 64, 3);
    wprep_kernel<<<(256 * 128 * 3 + 255) / 256, 256, 0, stream>>>(enc_w2, wE2, 256, 128, 3);
    wprep_kernel<<<(256 * 256 * 1 + 255) / 256, 256, 0, stream>>>(proj_w, wP, 256, 256, 1);
    wprep_kernel<<<(256 * 256 * 3 + 255) / 256, 256, 0, stream>>>(dec_w0, wD0, 256, 256, 3);
    wprep_kernel<<<(128 * 256 * 3 + 255) / 256, 256, 0, stream>>>(dec_w1, wD1, 128, 256, 3);
    wprep_kernel<<<(64 * 128 * 3 + 255) / 256, 256, 0, stream>>>(dec_w2, wD2, 64, 128, 3);
    wprep_kernel<<<(32 * 64 * 3 + 255) / 256, 256, 0, stream>>>(dec_w3, wD3, 32, 64, 3);

    // ---- encoder: stride-2 convs as 3-tap gathered GEMMs, ReLU ----------
    conv(xBF, wE0, enc_b0, nullptr, h0, 32, 64, 512, 256, 2, -1, 0, 1, 3, 0, 1, 2, 256, 1, 0, 1);
    conv(h0, wE1, enc_b1, nullptr, h1, 64, 128, 256, 128, 2, -1, 0, 1, 3, 0, 1, 2, 128, 1, 0, 1);
    conv(h1, wE2, enc_b2, nullptr, h2, 128, 256, 128, 64, 2, -1, 0, 1, 3, 0, 1, 2, 64, 1, 0, 1);
    // proj (k=1 GEMM): emits f32 z_e (for loss) + bf16 rows (VQ A operand)
    conv(h2, wP, proj_b, zeF, flatA, 256, 256, 64, 64, 1, 0, 0, 0, 1, 0, 0, 0, 64, 1, 0, 0);

    // ---- vector quantizer ----------------------------------------------
    vq_argmin_kernel<<<NROWS / 64, 128, 0, stream>>>(flatA, embBF, norms, idxB);
    gather_loss_kernel<<<NROWS * CH / 256, 256, 0, stream>>>(idxB, emb, zeF, zqBF, part);

    // ---- decoder: ConvTranspose split into even (W1) / odd (W0,W2) GEMMs
    conv(zqBF, wD0, dec_b0, nullptr, d0, 256, 256, 64, 64, 1, 0, 0, 0, 1, 1, 0, 0, 128, 2, 0, 1);
    conv(zqBF, wD0, dec_b0, nullptr, d0, 256, 256, 64, 64, 1, 0, 1, 0, 2, 0, 2, 0, 128, 2, 1, 1);
    conv(d0, wD1, dec_b1, nullptr, d1, 256, 128, 128, 128, 1, 0, 0, 0, 1, 1, 0, 0, 256, 2, 0, 1);
    conv(d0, wD1, dec_b1, nullptr, d1, 256, 128, 128, 128, 1, 0, 1, 0, 2, 0, 2, 0, 256, 2, 1, 1);
    conv(d1, wD2, dec_b2, nullptr, d2, 128, 64, 256, 256, 1, 0, 0, 0, 1, 1, 0, 0, 512, 2, 0, 1);
    conv(d1, wD2, dec_b2, nullptr, d2, 128, 64, 256, 256, 1, 0, 1, 0, 2, 0, 2, 0, 512, 2, 1, 1);
    // last layer: f32 straight into d_out; (B, L, A) layout == row-major rows x 32
    conv(d2, wD3, dec_b3, out, nullptr, 64, 32, 512, 512, 1, 0, 0, 0, 1, 1, 0, 0, 1024, 2, 0, 0);
    conv(d2, wD3, dec_b3, out, nullptr, 64, 32, 512, 512, 1, 0, 1, 0, 2, 0, 2, 0, 1024, 2, 1, 0);

    // ---- losses + indices ----------------------------------------------
    finalize_kernel<<<(NROWS + 255) / 256, 256, 0, stream>>>(part, idxB, outLoss, outIdx);
    (void)in_sizes; (void)n_in; (void)out_size; (void)ws_size;
}
